// MolGeomPredictionHead_54022098649981
// MI455X (gfx1250) — compile-verified, matrix operand order checked by reading, not run on hardware
//
#include <hip/hip_runtime.h>
#include <hip/hip_bf16.h>
#include <math.h>

// ---------------- problem constants (from reference) ----------------
#define NDIM   256
#define PDIM   128
#define NHEAD  8
#define DHEAD  16
#define NB_F   32
#define NI_    8192
#define NT_    16384
#define NUI_   49152
#define NUV_   65536
#define NNODE  (NT_ + NI_)        // 24576
#define NEDGE  (2*NUI_ + NUV_)    // 163840
#define EMB    (NDIM + 2*NB_F)    // 320

typedef __attribute__((ext_vector_type(16))) __bf16        v16bf;
typedef __attribute__((ext_vector_type(8)))  float         v8f;
typedef __attribute__((ext_vector_type(4)))  unsigned int  v4u;
typedef __attribute__((ext_vector_type(8)))  int           v8i;
typedef __attribute__((ext_vector_type(4)))  int           v4i;

#ifndef __has_builtin
#define __has_builtin(x) 0
#endif
#if __has_builtin(__builtin_amdgcn_tensor_load_to_lds) && \
    __has_builtin(__builtin_amdgcn_s_wait_tensorcnt)
#define USE_TDM 1
#else
#define USE_TDM 0
#endif

// All GEMMs in this model have M % 32 == 0 and K % 32 == 0 (asserted at the
// call sites below), so only N needs bounds handling (N in {8,128,192,320,384,512}).

#if USE_TDM
// Issue a TDM DMA of a 32-row x 32-float fp32 tile (row stride K floats) from
// global memory into LDS at byte offset lds_off, with LDS row padding of
// 4 DWORDs (row stride 36 floats -> keeps 16B alignment, no bank conflicts).
__device__ __forceinline__ void tdm_load_a32x32(const float* gptr, unsigned lds_off,
                                                int M, int K) {
    unsigned long long ga = (unsigned long long)(size_t)gptr;
    v4u g0;
    g0[0] = 1u;                                   // count=1, user descriptor
    g0[1] = lds_off;                              // lds_addr (bytes)
    g0[2] = (unsigned)ga;                         // global_addr[31:0]
    g0[3] = (unsigned)((ga >> 32) & 0x01FFFFFFull) | (2u << 30); // addr[56:32] | type=2
    v8i g1;
    // data_size=4B (2), pad_enable, pad_interval=32 DWORDs (4), pad_amount=4 DWORDs (3)
    g1[0] = (int)((2u << 16) | (1u << 20) | (4u << 22) | (3u << 25));
    g1[1] = (int)(((unsigned)K & 0xFFFFu) << 16);                   // tensor_dim0 lo
    g1[2] = (int)(((unsigned)K >> 16) | (((unsigned)M & 0xFFFFu) << 16)); // dim0 hi | dim1 lo
    g1[3] = (int)(((unsigned)M >> 16) | (32u << 16));               // dim1 hi | tile_dim0=32
    g1[4] = 32;                                                      // tile_dim1=32
    g1[5] = (int)K;                                                  // tensor_dim0_stride lo
    g1[6] = 0;
    g1[7] = 0;
    v4i z4; z4[0] = 0; z4[1] = 0; z4[2] = 0; z4[3] = 0;              // 2D: groups 2/3 unused
    v8i z8;
#pragma unroll
    for (int j = 0; j < 8; ++j) z8[j] = 0;
    // 6-arg form (clang-23 / therock-10.0 headers)
    __builtin_amdgcn_tensor_load_to_lds(g0, g1, z4, z4, z8, 0);
}
#endif

// ============================================================
// Tiled WMMA GEMM:  C[M,N] (=/+=) act(A[M,K] @ B[K,N] + bias)
// fp32 in HBM; A staged fp32 via TDM (double-buffered, TENSORcnt-pipelined),
// B staged bf16 (transposed) via vectorized loads; bf16 matrix pipes, f32 acc.
// Block tile 32(M) x 128(N), K-step 32, 8 waves, 2 wmma/wave/iter.
// ============================================================
__global__ __launch_bounds__(256)
void wmma_gemm(const float* __restrict__ A, const float* __restrict__ B,
               const float* __restrict__ bias, float* __restrict__ C,
               int M, int N, int K, int act, int accum)
{
    __shared__ __align__(16) float  sAf[2][32][36];   // fp32 A tiles (TDM pad stride 36)
    __shared__ __align__(16) __bf16 sB[128][36];      // bf16 B tile, [n][k] transposed

    const int tid  = threadIdx.x;
    const int lane = tid & 31;
    const int w    = tid >> 5;     // 0..7
    const int wm   = w & 1;        // 2 wave-rows
    const int wn   = w >> 1;       // 4 wave-cols, each covers 32 N
    const int m0   = blockIdx.y * 32;
    const int n0   = blockIdx.x * 128;
    const int fr   = lane & 15;
    const int koff = (lane >> 4) << 3;     // 0 or 8 (16-bit operand layout)
    const bool nfull = (n0 + 128 <= N);

    v8f acc[2];
#pragma unroll
    for (int r = 0; r < 8; ++r) { acc[0][r] = 0.f; acc[1][r] = 0.f; }

    int buf = 0;
#if USE_TDM
    if (w == 0)
        tdm_load_a32x32(A + (size_t)m0 * K, (unsigned)(size_t)&sAf[0][0][0], M, K);
#endif

    for (int k0 = 0; k0 < K; k0 += 32) {
        // ---- stage B tile (bf16, transposed [n][k]) ----
        if (nfull) {
            const int n4 = (tid & 31) << 2;
            for (int kk = tid >> 5; kk < 32; kk += 8) {
                float4 b4 = *(const float4*)(B + (size_t)(k0 + kk) * N + n0 + n4);
                sB[n4 + 0][kk] = (__bf16)b4.x;
                sB[n4 + 1][kk] = (__bf16)b4.y;
                sB[n4 + 2][kk] = (__bf16)b4.z;
                sB[n4 + 3][kk] = (__bf16)b4.w;
            }
        } else {
            for (int i = tid; i < 128 * 32; i += 256) {
                int nq = i >> 5, kk = i & 31;
                int gn = n0 + nq;
                int gnc = (gn < N) ? gn : (N - 1);       // clamped, branchless
                float bv = B[(size_t)(k0 + kk) * N + gnc];
                sB[nq][kk] = (__bf16)((gn < N) ? bv : 0.f);
            }
        }

        // ---- A tile ----
#if USE_TDM
        if (w == 0) {
            if (k0 + 32 < K) {   // prefetch next A tile into the other buffer
                tdm_load_a32x32(A + (size_t)m0 * K + (k0 + 32),
                                (unsigned)(size_t)&sAf[buf ^ 1][0][0], M, K);
                __builtin_amdgcn_s_wait_tensorcnt(1);  // current tile landed
            } else {
                __builtin_amdgcn_s_wait_tensorcnt(0);
            }
        }
#else
        {
            const int m  = tid >> 3;
            const int k4 = (tid & 7) << 2;
            float4 a4 = *(const float4*)(A + (size_t)(m0 + m) * K + k0 + k4);
            *(float4*)&sAf[0][m][k4] = a4;
        }
#endif
        __syncthreads();

        // ---- fragments + WMMA ----
        v16bf fa, fb0, fb1;
        const float* ar = &sAf[buf][wm * 16 + fr][0];
        const int bn0 = wn * 32 + fr;
        const int bn1 = bn0 + 16;
#pragma unroll
        for (int e = 0; e < 8; ++e) {
            fa[e]      = (__bf16)ar[koff + e];
            fa[8 + e]  = (__bf16)ar[16 + koff + e];
            fb0[e]     = sB[bn0][koff + e];
            fb0[8 + e] = sB[bn0][16 + koff + e];
            fb1[e]     = sB[bn1][koff + e];
            fb1[8 + e] = sB[bn1][16 + koff + e];
        }
        acc[0] = __builtin_amdgcn_wmma_f32_16x16x32_bf16(
                     false, fa, false, fb0, (short)0, acc[0], false, false);
        acc[1] = __builtin_amdgcn_wmma_f32_16x16x32_bf16(
                     false, fa, false, fb1, (short)0, acc[1], false, false);
        __syncthreads();
#if USE_TDM
        buf ^= 1;
#endif
    }

    // ---- epilogue: bias / GELU / residual-accumulate ----
    const int mrow = m0 + wm * 16 + ((lane >> 4) << 3);
#pragma unroll
    for (int t = 0; t < 2; ++t) {
        int col = n0 + wn * 32 + (lane & 15) + t * 16;
        if (col >= N) continue;
#pragma unroll
        for (int r = 0; r < 8; ++r) {
            int gm = mrow + r;                 // M % 32 == 0 at every call site
            float val = acc[t][r];
            if (bias) val += bias[col];
            if (act == 1) {                    // tanh-GELU (jax.nn.gelu default)
                float x = val;
                val = 0.5f * x * (1.f + tanhf(0.7978845608028654f *
                                              (x + 0.044715f * x * x * x)));
            }
            size_t o = (size_t)gm * N + col;
            if (accum) C[o] += val; else C[o] = val;
        }
    }
}

// ---------------- elementwise / gather kernels ----------------

__global__ void k_fill(float* p, long n, float v) {
    long i = (long)blockIdx.x * blockDim.x + threadIdx.x;
    if (i < n) p[i] = v;
}

__global__ void k_copyf(const float* s, float* d, long n) {
    long i = (long)blockIdx.x * blockDim.x + threadIdx.x;
    if (i < n) d[i] = s[i];
}

__global__ void k_copy_cols(const float* __restrict__ src, float* __restrict__ dst,
                            long rows, int wsrc, int wdst, int off) {
    long i = (long)blockIdx.x * blockDim.x + threadIdx.x;
    if (i >= rows * wsrc) return;
    long r = i / wsrc; int c = (int)(i % wsrc);
    dst[r * (long)wdst + off + c] = src[i];
}

__global__ void k_build_hin(const float* __restrict__ te, const int* __restrict__ molid,
                            const float* __restrict__ attr, const float* __restrict__ fourier,
                            float* __restrict__ hin) {
    long idx = (long)blockIdx.x * blockDim.x + threadIdx.x;
    if (idx >= (long)NI_ * EMB) return;
    int i = (int)(idx / EMB), j = (int)(idx % EMB);
    float v;
    if (j < NDIM) v = attr[(size_t)i * NDIM + j];
    else {
        float tm = te[molid[i]];
        int b = j - NDIM;
        float ang = 6.283185307179586f * tm *
                    fourier[(b < NB_F) ? b : (b - NB_F)];
        v = (b < NB_F) ? sinf(ang) : cosf(ang);
    }
    hin[idx] = v;
}

__global__ void k_frames(const float* __restrict__ coords,
                         const int* __restrict__ ijk_i, const int* __restrict__ ijk_j,
                         const int* __restrict__ ijk_k, const int* __restrict__ I_ijk,
                         float* __restrict__ R, float* __restrict__ T) {
    int i = blockIdx.x * blockDim.x + threadIdx.x;
    if (i >= NNODE) return;
    float* Ri = R + (size_t)i * 9;
    float* ti = T + (size_t)i * 3;
    if (i < NT_) {
        int tr = I_ijk[i];
        int a = ijk_i[tr], b = ijk_j[tr], c = ijk_k[tr];
        float x1[3], x2[3], x3[3], v1[3], v2[3], e1[3], u2[3], e2[3], e3[3];
#pragma unroll
        for (int d = 0; d < 3; ++d) {
            x1[d] = coords[(size_t)a * 3 + d];
            x2[d] = coords[(size_t)b * 3 + d];
            x3[d] = coords[(size_t)c * 3 + d];
            v1[d] = x1[d] - x2[d];
            v2[d] = x3[d] - x2[d];
        }
        float r1 = rsqrtf(v1[0]*v1[0] + v1[1]*v1[1] + v1[2]*v1[2] + 1e-8f);
#pragma unroll
        for (int d = 0; d < 3; ++d) e1[d] = v1[d] * r1;
        float dp = e1[0]*v2[0] + e1[1]*v2[1] + e1[2]*v2[2];
#pragma unroll
        for (int d = 0; d < 3; ++d) u2[d] = v2[d] - e1[d] * dp;
        float r2 = rsqrtf(u2[0]*u2[0] + u2[1]*u2[1] + u2[2]*u2[2] + 1e-8f);
#pragma unroll
        for (int d = 0; d < 3; ++d) e2[d] = u2[d] * r2;
        e3[0] = e1[1]*e2[2] - e1[2]*e2[1];
        e3[1] = e1[2]*e2[0] - e1[0]*e2[2];
        e3[2] = e1[0]*e2[1] - e1[1]*e2[0];
        // R[i][j] = e_j[i]  (columns are e1,e2,e3)
#pragma unroll
        for (int d = 0; d < 3; ++d) {
            Ri[d*3+0] = e1[d]; Ri[d*3+1] = e2[d]; Ri[d*3+2] = e3[d];
            ti[d] = x2[d];
        }
    } else {
        int u = i - NT_;
#pragma unroll
        for (int d = 0; d < 9; ++d) Ri[d] = 0.f;
        Ri[0] = Ri[4] = Ri[8] = 1.f;
#pragma unroll
        for (int d = 0; d < 3; ++d) ti[d] = coords[(size_t)u * 3 + d];
    }
}

__global__ void k_rotate_trpv(const float* __restrict__ trpv, const float* __restrict__ R,
                              float* __restrict__ vbuf) {
    long idx = (long)blockIdx.x * blockDim.x + threadIdx.x;
    if (idx >= (long)NT_ * PDIM) return;
    int I = (int)(idx / PDIM), d = (int)(idx % PDIM);
    const float* Ri = R + (size_t)I * 9;
    float t0 = trpv[(size_t)I * 384 + d];
    float t1 = trpv[(size_t)I * 384 + 128 + d];
    float t2 = trpv[(size_t)I * 384 + 256 + d];
#pragma unroll
    for (int c = 0; c < 3; ++c)
        vbuf[((size_t)I * 3 + c) * PDIM + d] =
            Ri[c*3+0]*t0 + Ri[c*3+1]*t1 + Ri[c*3+2]*t2;
}

// one wave per row of 128
__global__ __launch_bounds__(256)
void k_layernorm(const float* __restrict__ x, const float* __restrict__ g,
                 const float* __restrict__ b, float* __restrict__ y, int n) {
    int row  = blockIdx.x * 8 + (threadIdx.x >> 5);
    int lane = threadIdx.x & 31;
    if (row >= n) return;
    const float* xr = x + (size_t)row * PDIM;
    float s = 0.f, s2 = 0.f;
    for (int i = lane; i < PDIM; i += 32) { float v = xr[i]; s += v; s2 += v * v; }
#pragma unroll
    for (int o = 16; o > 0; o >>= 1) {
        s  += __shfl_xor(s,  o, 32);
        s2 += __shfl_xor(s2, o, 32);
    }
    float m = s * (1.f / PDIM);
    float var = s2 * (1.f / PDIM) - m * m;
    float r = rsqrtf(var + 1e-5f);
    for (int i = lane; i < PDIM; i += 32)
        y[(size_t)row * PDIM + i] = (xr[i] - m) * r * g[i] + b[i];
}

// local points -> global frame: out = R*loc + t  (per node, per (h,p))
__global__ void k_transform_pts(const float* __restrict__ loc, const float* __restrict__ R,
                                const float* __restrict__ T, float* __restrict__ out) {
    long idx = (long)blockIdx.x * blockDim.x + threadIdx.x;
    if (idx >= (long)NNODE * 64) return;
    int i = (int)(idx >> 6), hp = (int)(idx & 63);
    const float* l  = loc + (size_t)i * 192 + hp * 3;
    const float* Ri = R + (size_t)i * 9;
    const float* ti = T + (size_t)i * 3;
#pragma unroll
    for (int c = 0; c < 3; ++c)
        out[(size_t)i * 192 + hp * 3 + c] =
            Ri[c*3+0]*l[0] + Ri[c*3+1]*l[1] + Ri[c*3+2]*l[2] + ti[c];
}

__device__ __forceinline__ void edge_decode(int e, const int* UI_I, const int* UI_u,
                                            const int* uv_u, const int* uv_v,
                                            int& src, int& dst, int& kind, int& row) {
    if (e < NUI_)            { src = UI_I[e];            dst = NT_ + UI_u[e];      kind = 0; row = e; }
    else if (e < 2 * NUI_)   { int r = e - NUI_;
                               src = NT_ + UI_u[r];      dst = UI_I[r];            kind = 0; row = r; }
    else                     { int r = e - 2 * NUI_;
                               src = NT_ + uv_u[r];      dst = NT_ + uv_v[r];      kind = 1; row = r; }
}

__device__ __forceinline__ void atomicMaxOrdered(unsigned int* a, float v) {
    unsigned int nv = __float_as_uint(v);
    unsigned int enc = (nv & 0x80000000u) ? ~nv : (nv | 0x80000000u);
    atomicMax(a, enc);
}

__global__ void k_edge_logits(const float* __restrict__ q, const float* __restrict__ k,
                              const float* __restrict__ qp, const float* __restrict__ kp,
                              const float* __restrict__ bias_af, const float* __restrict__ bias_uv,
                              const float* __restrict__ gamma_p,
                              const int* UI_I, const int* UI_u, const int* uv_u, const int* uv_v,
                              float* __restrict__ logits, unsigned int* __restrict__ segmax) {
    long idx = (long)blockIdx.x * blockDim.x + threadIdx.x;
    if (idx >= (long)NEDGE * NHEAD) return;
    int e = (int)(idx >> 3), h = (int)(idx & 7);
    int src, dst, kind, row;
    edge_decode(e, UI_I, UI_u, uv_u, uv_v, src, dst, kind, row);
    const float* qd = q + (size_t)dst * PDIM + h * DHEAD;
    const float* ks = k + (size_t)src * PDIM + h * DHEAD;
    float lg = 0.f;
#pragma unroll
    for (int d = 0; d < DHEAD; ++d) lg += qd[d] * ks[d];
    lg *= 0.25f;                                   // 1/sqrt(16)
    lg += (kind ? bias_uv : bias_af)[(size_t)row * NHEAD + h];
    float g = gamma_p[h];
    float sp = (g > 20.f) ? g : log1pf(expf(g));   // softplus
    const float* qpp = qp + (size_t)dst * 192 + h * 24;
    const float* kpp = kp + (size_t)src * 192 + h * 24;
    float d2 = 0.f;
#pragma unroll
    for (int d = 0; d < 24; ++d) { float df = qpp[d] - kpp[d]; d2 += df * df; }
    lg -= 0.5f * sp * d2;
    logits[idx] = lg;
    atomicMaxOrdered(&segmax[(size_t)dst * NHEAD + h], lg);
}

__global__ void k_edge_p(float* __restrict__ logits, const unsigned int* __restrict__ segmax,
                         const int* UI_I, const int* UI_u, const int* uv_u, const int* uv_v,
                         float* __restrict__ segsum) {
    long idx = (long)blockIdx.x * blockDim.x + threadIdx.x;
    if (idx >= (long)NEDGE * NHEAD) return;
    int e = (int)(idx >> 3), h = (int)(idx & 7);
    int src, dst, kind, row;
    edge_decode(e, UI_I, UI_u, uv_u, uv_v, src, dst, kind, row);
    unsigned int enc = segmax[(size_t)dst * NHEAD + h];
    float m = 0.f;
    if (enc != 0u) {
        unsigned int nv = (enc & 0x80000000u) ? (enc & 0x7fffffffu) : ~enc;
        m = __uint_as_float(nv);
        if (!(fabsf(m) < 3.0e38f)) m = 0.f;        // non-finite -> 0 (matches ref)
    }
    float pv = expf(logits[idx] - m);
    logits[idx] = pv;                              // overwrite with unnormalized p
    atomicAdd(&segsum[(size_t)dst * NHEAD + h], pv);
}

__global__ void k_edge_scatter(const float* __restrict__ p_eh, const float* __restrict__ segsum,
                               const int* UI_I, const int* UI_u, const int* uv_u, const int* uv_v,
                               const float* __restrict__ vv,
                               const float* __restrict__ ve_af, const float* __restrict__ ve_uv,
                               const float* __restrict__ vpg, const float* __restrict__ vval,
                               float* __restrict__ o_s, float* __restrict__ o_e,
                               float* __restrict__ o_pg, float* __restrict__ o_vec) {
    long idx = (long)blockIdx.x * blockDim.x + threadIdx.x;
    if (idx >= (long)NEDGE * NHEAD) return;
    int e = (int)(idx >> 3), h = (int)(idx & 7);
    int src, dst, kind, row;
    edge_decode(e, UI_I, UI_u, uv_u, uv_v, src, dst, kind, row);
    float a = p_eh[idx] / (segsum[(size_t)dst * NHEAD + h] + 1e-9f);

    const float* vs = vv + (size_t)src * PDIM + h * DHEAD;
    float* os = o_s + (size_t)dst * PDIM + h * DHEAD;
#pragma unroll
    for (int d = 0; d < DHEAD; ++d) atomicAdd(&os[d], a * vs[d]);

    const float* ver = (kind ? ve_uv : ve_af) + (size_t)row * PDIM + h * DHEAD;
    float* oe = o_e + (size_t)dst * PDIM + h * DHEAD;
#pragma unroll
    for (int d = 0; d < DHEAD; ++d) atomicAdd(&oe[d], a * ver[d]);

    const float* vp = vpg + (size_t)src * 192 + h * 24;
    float* op = o_pg + (size_t)dst * 192 + h * 24;
#pragma unroll
    for (int d = 0; d < 24; ++d) atomicAdd(&op[d], a * vp[d]);

#pragma unroll
    for (int ch = 0; ch < 3; ++ch) {
        const float* vvs = vval + ((size_t)src * 3 + ch) * PDIM + h * DHEAD;
        float* ov = o_vec + ((size_t)dst * 3 + ch) * PDIM + h * DHEAD;
#pragma unroll
        for (int d = 0; d < DHEAD; ++d) atomicAdd(&ov[d], a * vvs[d]);
    }
}

// rel = o_pg - t; o_pl = R^T rel; o_pn = |o_pl|; fill sc[256:512] and vcat[128:192]
__global__ void k_node_post(const float* __restrict__ o_pg, const float* __restrict__ R,
                            const float* __restrict__ T, float* __restrict__ sc,
                            float* __restrict__ vcat) {
    long idx = (long)blockIdx.x * blockDim.x + threadIdx.x;
    if (idx >= (long)NNODE * 64) return;
    int i = (int)(idx >> 6), hp = (int)(idx & 63);
    const float* Ri = R + (size_t)i * 9;
    const float* ti = T + (size_t)i * 3;
    float rel[3], pl[3];
#pragma unroll
    for (int c = 0; c < 3; ++c) rel[c] = o_pg[(size_t)i * 192 + hp * 3 + c] - ti[c];
#pragma unroll
    for (int j = 0; j < 3; ++j)
        pl[j] = Ri[j] * rel[0] + Ri[3 + j] * rel[1] + Ri[6 + j] * rel[2];
    float pn = sqrtf(pl[0]*pl[0] + pl[1]*pl[1] + pl[2]*pl[2] + 1e-8f);
    float* sci = sc + (size_t)i * 512;
    sci[256 + hp * 3 + 0] = pl[0];
    sci[256 + hp * 3 + 1] = pl[1];
    sci[256 + hp * 3 + 2] = pl[2];
    sci[448 + hp] = pn;
#pragma unroll
    for (int ch = 0; ch < 3; ++ch)
        vcat[((size_t)i * 3 + ch) * 192 + 128 + hp] = rel[ch];
}

__global__ void k_coord_update(const float* __restrict__ s, const float* __restrict__ v,
                               const float* __restrict__ Wsc, const float* __restrict__ bsc,
                               const float* __restrict__ Watm, float* __restrict__ coords) {
    int u = blockIdx.x * blockDim.x + threadIdx.x;
    if (u >= NI_) return;
    const float* sr = s + (size_t)(NT_ + u) * PDIM;
    float z = bsc[0];
    for (int d = 0; d < PDIM; ++d) z += sr[d] * Wsc[d];
    float scale = 10.f / (1.f + expf(-z));
#pragma unroll
    for (int ch = 0; ch < 3; ++ch) {
        const float* vr = v + ((size_t)(NT_ + u) * 3 + ch) * PDIM;
        float dd = 0.f;
        for (int d = 0; d < PDIM; ++d) dd += vr[d] * Watm[d];
        coords[(size_t)u * 3 + ch] += dd * scale;
    }
}

// ---------------- host-side orchestration ----------------
static inline void gemm(hipStream_t s, const float* A, const float* B, const float* bias,
                        float* C, int M, int N, int K, int act, int accum) {
    // invariant: M % 32 == 0 && K % 32 == 0 for every call in this model
    dim3 g((N + 127) / 128, (M + 31) / 32);
    wmma_gemm<<<g, 256, 0, s>>>(A, B, bias, C, M, N, K, act, accum);
}
static inline void fill(hipStream_t s, float* p, long n, float v) {
    k_fill<<<(unsigned)((n + 255) / 256), 256, 0, s>>>(p, n, v);
}

struct Blk { // JAX pytree (sorted key) order inside each block dict
    const float *Wb, *Wf1, *Wf2, *Wk, *Wkp, *Wo, *Wq, *Wqp, *Wv, *Wve, *Wvo, *Wvp,
                *Wvv, *bf1, *bf2, *bo, *gamma, *ln1_b, *ln1_g, *ln2_b, *ln2_g;
};

extern "C" void kernel_launch(void* const* d_in, const int* in_sizes, int n_in,
                              void* d_out, int out_size, void* d_ws, size_t ws_size,
                              hipStream_t stream) {
    (void)in_sizes; (void)n_in; (void)out_size; (void)ws_size;

    const float* te        = (const float*)d_in[0];
    const float* coords_in = (const float*)d_in[1];
    const float* atom_attr = (const float*)d_in[2];
    const float* trp_attr  = (const float*)d_in[3];
    const float* af_pair   = (const float*)d_in[4];
    const float* uv_pair   = (const float*)d_in[5];
    const int*   molid     = (const int*)d_in[6];
    const int*   ijk_i     = (const int*)d_in[7];
    const int*   ijk_j     = (const int*)d_in[8];
    const int*   ijk_k     = (const int*)d_in[9];
    const int*   I_ijk     = (const int*)d_in[10];
    const int*   UI_I      = (const int*)d_in[11];
    const int*   UI_u      = (const int*)d_in[12];
    const int*   uv_u      = (const int*)d_in[13];
    const int*   uv_v      = (const int*)d_in[14];
    // params flattened in sorted-pytree order:
    const float* Watm  = (const float*)d_in[15];
    const float* We1   = (const float*)d_in[16];
    const float* We2   = (const float*)d_in[17];
    const float* Wsc   = (const float*)d_in[18];
    const float* Wtrp  = (const float*)d_in[19];
    const float* Wtrpv = (const float*)d_in[20];
    const float* be1   = (const float*)d_in[21];
    const float* be2   = (const float*)d_in[22];
    Blk blk[3];
    for (int b = 0; b < 3; ++b) {
        const float** f = (const float**)&blk[b];
        for (int j = 0; j < 21; ++j) f[j] = (const float*)d_in[23 + b * 21 + j];
    }
    const float* bsc     = (const float*)d_in[86];
    const float* fourier = (const float*)d_in[87];

    // bump allocator over workspace
    char* ws = (char*)d_ws;
    size_t off = 0;
    auto alloc = [&](size_t nf) -> float* {
        float* p = (float*)(ws + off);
        off += ((nf * sizeof(float) + 255) / 256) * 256;
        return p;
    };
    float* coords = alloc((size_t)NI_ * 3);
    float* hin    = alloc((size_t)NI_ * EMB);
    float* t320   = alloc((size_t)NI_ * EMB);
    float* sbuf   = alloc((size_t)NNODE * PDIM);
    float* vbuf   = alloc((size_t)NNODE * 3 * PDIM);
    float* hn     = alloc((size_t)NNODE * PDIM);
    float* Rbuf   = alloc((size_t)NNODE * 9);
    float* Tbuf   = alloc((size_t)NNODE * 3);
    float* qb     = alloc((size_t)NNODE * PDIM);
    float* kb     = alloc((size_t)NNODE * PDIM);
    float* vvb    = alloc((size_t)NNODE * PDIM);
    float* ploc   = alloc((size_t)NNODE * 192);
    float* qp     = alloc((size_t)NNODE * 192);
    float* kp     = alloc((size_t)NNODE * 192);
    float* vp     = alloc((size_t)NNODE * 192);
    float* trpv3  = alloc((size_t)NT_ * 384);
    float* bias_af= alloc((size_t)NUI_ * NHEAD);
    float* bias_uv= alloc((size_t)NUV_ * NHEAD);
    float* ve_af  = alloc((size_t)NUI_ * PDIM);
    float* ve_uv  = alloc((size_t)NUV_ * PDIM);
    float* logit  = alloc((size_t)NEDGE * NHEAD);
    float* segmax = alloc((size_t)NNODE * NHEAD);
    float* segsum = alloc((size_t)NNODE * NHEAD);
    float* o_s    = alloc((size_t)NNODE * PDIM);
    float* o_e    = alloc((size_t)NNODE * PDIM);
    float* o_pg   = alloc((size_t)NNODE * 192);
    float* o_vec  = alloc((size_t)NNODE * 3 * PDIM);
    float* vval   = alloc((size_t)NNODE * 3 * PDIM);
    float* scb    = alloc((size_t)NNODE * 512);
    float* vcat   = alloc((size_t)NNODE * 576);
    float* ffn    = alloc((size_t)NNODE * 512);

    const long EH = (long)NEDGE * NHEAD;
    const unsigned gEH = (unsigned)((EH + 255) / 256);
    const unsigned gHP = (unsigned)(((long)NNODE * 64 + 255) / 256);

    // coords working copy
    k_copyf<<<(NI_ * 3 + 255) / 256, 256, 0, stream>>>(coords_in, coords, (long)NI_ * 3);

    // ---- atom embedding MLP (WMMA) ----
    k_build_hin<<<(unsigned)(((long)NI_ * EMB + 255) / 256), 256, 0, stream>>>(
        te, molid, atom_attr, fourier, hin);
    gemm(stream, hin, We1, be1, t320, NI_, EMB, EMB, /*gelu*/1, 0);
    gemm(stream, t320, We2, be2, sbuf + (size_t)NT_ * PDIM, NI_, PDIM, EMB, 0, 0);

    // ---- trp init (frames from initial coords) ----
    k_frames<<<(NNODE + 255) / 256, 256, 0, stream>>>(coords, ijk_i, ijk_j, ijk_k, I_ijk, Rbuf, Tbuf);
    gemm(stream, trp_attr, Wtrp, nullptr, sbuf, NT_, PDIM, NDIM, 0, 0);
    gemm(stream, trp_attr, Wtrpv, nullptr, trpv3, NT_, 384, NDIM, 0, 0);
    fill(stream, vbuf, (long)NNODE * 3 * PDIM, 0.f);
    k_rotate_trpv<<<(unsigned)(((long)NT_ * PDIM + 255) / 256), 256, 0, stream>>>(trpv3, Rbuf, vbuf);

    // ---- 3 IPA blocks ----
    for (int b = 0; b < 3; ++b) {
        const Blk& B = blk[b];
        k_frames<<<(NNODE + 255) / 256, 256, 0, stream>>>(coords, ijk_i, ijk_j, ijk_k, I_ijk, Rbuf, Tbuf);

        // per-edge projections of pair features
        gemm(stream, af_pair, B.Wb,  nullptr, bias_af, NUI_, NHEAD, PDIM, 0, 0);
        gemm(stream, uv_pair, B.Wb,  nullptr, bias_uv, NUV_, NHEAD, PDIM, 0, 0);
        gemm(stream, af_pair, B.Wve, nullptr, ve_af,   NUI_, PDIM,  PDIM, 0, 0);
        gemm(stream, uv_pair, B.Wve, nullptr, ve_uv,   NUV_, PDIM,  PDIM, 0, 0);

        // node projections
        k_layernorm<<<(NNODE + 7) / 8, 256, 0, stream>>>(sbuf, B.ln1_g, B.ln1_b, hn, NNODE);
        gemm(stream, hn, B.Wq, nullptr, qb,  NNODE, PDIM, PDIM, 0, 0);
        gemm(stream, hn, B.Wk, nullptr, kb,  NNODE, PDIM, PDIM, 0, 0);
        gemm(stream, hn, B.Wv, nullptr, vvb, NNODE, PDIM, PDIM, 0, 0);
        gemm(stream, hn, B.Wqp, nullptr, ploc, NNODE, 192, PDIM, 0, 0);
        k_transform_pts<<<gHP, 256, 0, stream>>>(ploc, Rbuf, Tbuf, qp);
        gemm(stream, hn, B.Wkp, nullptr, ploc, NNODE, 192, PDIM, 0, 0);
        k_transform_pts<<<gHP, 256, 0, stream>>>(ploc, Rbuf, Tbuf, kp);
        gemm(stream, hn, B.Wvp, nullptr, ploc, NNODE, 192, PDIM, 0, 0);
        k_transform_pts<<<gHP, 256, 0, stream>>>(ploc, Rbuf, Tbuf, vp);
        gemm(stream, vbuf, B.Wvv, nullptr, vval, NNODE * 3, PDIM, PDIM, 0, 0);

        // segment softmax over edges
        fill(stream, segmax, (long)NNODE * NHEAD, 0.f);   // 0 bits == ordered -inf sentinel
        fill(stream, segsum, (long)NNODE * NHEAD, 0.f);
        fill(stream, o_s,  (long)NNODE * PDIM, 0.f);
        fill(stream, o_e,  (long)NNODE * PDIM, 0.f);
        fill(stream, o_pg, (long)NNODE * 192, 0.f);
        fill(stream, o_vec,(long)NNODE * 3 * PDIM, 0.f);
        k_edge_logits<<<gEH, 256, 0, stream>>>(qb, kb, qp, kp, bias_af, bias_uv, B.gamma,
                                               UI_I, UI_u, uv_u, uv_v,
                                               logit, (unsigned int*)segmax);
        k_edge_p<<<gEH, 256, 0, stream>>>(logit, (const unsigned int*)segmax,
                                          UI_I, UI_u, uv_u, uv_v, segsum);
        k_edge_scatter<<<gEH, 256, 0, stream>>>(logit, segsum, UI_I, UI_u, uv_u, uv_v,
                                                vvb, ve_af, ve_uv, vp, vval,
                                                o_s, o_e, o_pg, o_vec);

        // assemble sc / vcat, frame-inverse point features
        k_copy_cols<<<(unsigned)(((long)NNODE * PDIM + 255) / 256), 256, 0, stream>>>(
            o_s, scb, NNODE, PDIM, 512, 0);
        k_copy_cols<<<(unsigned)(((long)NNODE * PDIM + 255) / 256), 256, 0, stream>>>(
            o_e, scb, NNODE, PDIM, 512, 128);
        k_copy_cols<<<(unsigned)(((long)NNODE * 3 * PDIM + 255) / 256), 256, 0, stream>>>(
            o_vec, vcat, (long)NNODE * 3, PDIM, 192, 0);
        k_node_post<<<gHP, 256, 0, stream>>>(o_pg, Rbuf, Tbuf, scb, vcat);

        // residual updates (WMMA with fused accumulate)
        gemm(stream, vcat, B.Wvo, nullptr, vbuf, NNODE * 3, PDIM, 192, 0, 1); // v += ...
        gemm(stream, scb,  B.Wo,  B.bo,   sbuf, NNODE, PDIM, 512, 0, 1);      // s += ...

        // FFN
        k_layernorm<<<(NNODE + 7) / 8, 256, 0, stream>>>(sbuf, B.ln2_g, B.ln2_b, hn, NNODE);
        gemm(stream, hn,  B.Wf1, B.bf1, ffn,  NNODE, 512, PDIM, /*gelu*/1, 0);
        gemm(stream, ffn, B.Wf2, B.bf2, sbuf, NNODE, PDIM, 512, 0, 1);        // s += ...

        // coordinate update
        k_coord_update<<<(NI_ + 255) / 256, 256, 0, stream>>>(sbuf, vbuf, Wsc, bsc, Watm, coords);
    }

    k_copyf<<<(NI_ * 3 + 255) / 256, 256, 0, stream>>>(coords, (float*)d_out, (long)NI_ * 3);
}
